// Spatom_20521353740694
// MI455X (gfx1250) — matrix operand order repaired and checked by reading, not compiled, not run on hardware
//
#include <hip/hip_runtime.h>
#include <hip/hip_bf16.h>
#include <cmath>

// ---------------------------------------------------------------------------
// Problem constants (from reference)
// ---------------------------------------------------------------------------
#define NN     8192
#define NFEAT  63
#define NHID   1024
#define NLAYERS 7
#define ALPHA  0.7f
#define NEGV   -1.0e12f

typedef __attribute__((ext_vector_type(16))) __bf16 v16bf;
typedef __attribute__((ext_vector_type(8)))  float  v8f;

// ---------------------------------------------------------------------------
// bf16 helpers
// ---------------------------------------------------------------------------
__device__ __forceinline__ unsigned short f2bf(float f) {
    unsigned int u = __float_as_uint(f);
    u += 0x7FFFu + ((u >> 16) & 1u);   // round-to-nearest-even
    return (unsigned short)(u >> 16);
}
__device__ __forceinline__ uint2 pack4bf(float4 v) {
    uint2 r;
    r.x = (unsigned)f2bf(v.x) | ((unsigned)f2bf(v.y) << 16);
    r.y = (unsigned)f2bf(v.z) | ((unsigned)f2bf(v.w) << 16);
    return r;
}

// ---------------------------------------------------------------------------
// CDNA5 async copy: global -> LDS, 16B per lane, tracked with ASYNCcnt.
// Low 32 bits of a flat LDS pointer are the wave-relative LDS offset.
// ---------------------------------------------------------------------------
__device__ __forceinline__ void async_ld_b128(unsigned lds_off, const void* gsrc) {
    asm volatile("global_load_async_to_lds_b128 %0, %1, off"
                 :: "v"(lds_off), "v"(gsrc) : "memory");
}
__device__ __forceinline__ void wait_asynccnt0() {
    asm volatile("s_wait_asynccnt 0x0" ::: "memory");
}

// ---------------------------------------------------------------------------
// fp32 -> bf16 converts
// ---------------------------------------------------------------------------
// vectorized: n multiple of 4
__global__ void k_f32_to_bf16v(const float4* __restrict__ src,
                               uint2* __restrict__ dst, long n4) {
    long i = (long)blockIdx.x * blockDim.x + threadIdx.x;
    long stride = (long)gridDim.x * blockDim.x;
    for (; i < n4; i += stride) dst[i] = pack4bf(src[i]);
}

// transposed+padded convert: dst[n*K + k] = bf16(src[k*srcStride + n]) for
// k < kLim && n < nLim, else 0.   dst is N x K (BT layout), i over N*K.
__global__ void k_convT(const float* __restrict__ src, unsigned short* __restrict__ dst,
                        int K, int N, int srcStride, int kLim, int nLim) {
    long total = (long)N * K;
    long i = (long)blockIdx.x * blockDim.x + threadIdx.x;
    long stride = (long)gridDim.x * blockDim.x;
    for (; i < total; i += stride) {
        int n = (int)(i / K), k = (int)(i % K);
        float v = (k < kLim && n < nLim) ? src[(long)k * srcStride + n] : 0.f;
        dst[i] = f2bf(v);
    }
}

// x (8192 x 63 f32) -> xp (8192 x 64 bf16), last column zero  (A operand)
__global__ void k_pad_x(const float* __restrict__ x,
                        unsigned short* __restrict__ xp) {
    long i = (long)blockIdx.x * blockDim.x + threadIdx.x;
    long stride = (long)gridDim.x * blockDim.x;
    long n = (long)NN * 64;
    for (; i < n; i += stride) {
        long row = i >> 6; int col = (int)(i & 63);
        float v = (col < NFEAT) ? x[row * NFEAT + col] : 0.f;
        xp[i] = f2bf(v);
    }
}

// cls1_b (64) -> padded fp32 (128)
__global__ void k_pad_bias(const float* __restrict__ b, float* __restrict__ bp) {
    int i = threadIdx.x;                 // launched with 128 threads
    bp[i] = (i < 64) ? b[i] : 0.f;
}

// ---------------------------------------------------------------------------
// bf16 tiled transpose: dst[C][R] = src[R][C]   (R, C multiples of 32)
// Paid once per layer for h, instead of 32x inside the GEMM grid.
// ---------------------------------------------------------------------------
__global__ __launch_bounds__(256)
void k_transpose_bf16(const unsigned short* __restrict__ src,
                      unsigned short* __restrict__ dst, int R, int C) {
    __shared__ unsigned short tile[32][34];      // +2 pad: no bank conflicts
    const int tx = threadIdx.x & 31, ty = threadIdx.x >> 5;   // 32 x 8
    const long r0 = (long)blockIdx.y * 32, c0 = (long)blockIdx.x * 32;
#pragma unroll
    for (int i = ty; i < 32; i += 8)
        tile[i][tx] = src[(r0 + i) * C + c0 + tx];
    __syncthreads();
#pragma unroll
    for (int i = ty; i < 32; i += 8)
        dst[(c0 + i) * R + r0 + tx] = tile[tx][i];
}

// ---------------------------------------------------------------------------
// bf16 WMMA GEMM:  C[MxN,f32] = A[MxK,bf16] @ B, with B given TRANSPOSED as
// BT[NxK,bf16] (both row-major; K mult of 32, M mult of 256, N mult of 128).
//   EPI 0: C = acc          EPI 1: C = relu(acc + bias[n])
// 256 threads (8 waves), block tile 256x128, K-step 32, double-buffered LDS.
// Wave grid 4(M) x 2(N): each wave owns a 64x64 patch = 4x4 WMMA fragments,
// so both A and B fragments are reused 4x: 16 ds_load_b128 per 16 WMMAs.
// Both tiles are staged entirely with GLOBAL_LOAD_ASYNC_TO_LDS_B128 (no ds
// stores, no VGPR round-trip); staging of tile k+1 overlaps the 16 WMMAs of
// tile k. Fragment reads are contiguous ds_load_b128 pairs matching the
// CDNA5 16-bit A/B VGPR layouts.
// ---------------------------------------------------------------------------
#define BM 256
#define BN 128
#define BK 32

template <int EPI>
__global__ __launch_bounds__(256)
void k_gemm_bf16(const unsigned short* __restrict__ A,
                 const unsigned short* __restrict__ BT,
                 float* __restrict__ C,
                 int M, int N, int K,
                 const float* __restrict__ bias) {
    __shared__ alignas(16) unsigned short As[2][BM][BK];   // 32 KB
    __shared__ alignas(16) unsigned short Bs[2][BN][BK];   // 16 KB, [n][k]

    const int tid  = threadIdx.x;
    const int wave = tid >> 5;
    const int lane = tid & 31;
    const int half = lane >> 4;     // 0: lanes 0-15, 1: lanes 16-31
    const int lrow = lane & 15;
    const int wm   = (wave & 3) * 64;   // 4 waves along M
    const int wn   = (wave >> 2) * 64;  // 2 waves along N
    const long blockM = (long)blockIdx.y * BM;
    const long blockN = (long)blockIdx.x * BN;

    v8f acc[4][4];
#pragma unroll
    for (int i = 0; i < 4; i++)
#pragma unroll
        for (int j = 0; j < 4; j++)
#pragma unroll
            for (int e = 0; e < 8; e++) acc[i][j][e] = 0.f;

    // cooperative staging coords:
    //   A tile 256x32: one 64B row per thread  (4 async b128)
    //   B tile 128x32: half a row per thread   (2 async b128)
    const int br = tid >> 1;
    const int bc = (tid & 1) * 16;
    const unsigned short* aSrc = &A [(blockM + tid) * (long)K];
    const unsigned short* bSrc = &BT[(blockN + br) * (long)K + bc];

    auto stage = [&](int b, int kb) {
        unsigned da = (unsigned)(unsigned long long)&As[b][tid][0];
        async_ld_b128(da,      aSrc + kb);
        async_ld_b128(da + 16, aSrc + kb + 8);
        async_ld_b128(da + 32, aSrc + kb + 16);
        async_ld_b128(da + 48, aSrc + kb + 24);
        unsigned db = (unsigned)(unsigned long long)&Bs[b][br][bc];
        async_ld_b128(db,      bSrc + kb);
        async_ld_b128(db + 16, bSrc + kb + 8);
    };

    stage(0, 0);
    wait_asynccnt0();
    __syncthreads();

    int buf = 0;
    for (int kb = 0; kb < K; kb += BK) {
        if (kb + BK < K) {              // stage next K-tile into other buffer
            stage(buf ^ 1, kb + BK);
            if (kb + 2 * BK < K) {      // global_prefetch_b8 for tile after next
                __builtin_prefetch(aSrc + kb + 2 * BK, 0, 3);
                __builtin_prefetch(bSrc + kb + 2 * BK, 0, 3);
            }
        }

        // ---- compute on current buffer: 16 ds_load_b128 -> 16 WMMA ----
        v16bf afrag[4], bfrag[4];
#pragma unroll
        for (int mt = 0; mt < 4; mt++) {
            int row = wm + mt * 16 + lrow;
            uint4 pair[2];
            pair[0] = *(const uint4*)&As[buf][row][half * 8];        // K 0..7 / 8..15
            pair[1] = *(const uint4*)&As[buf][row][16 + half * 8];   // K 16..23 / 24..31
            __builtin_memcpy(&afrag[mt], pair, 32);
        }
#pragma unroll
        for (int nt = 0; nt < 4; nt++) {
            int col = wn + nt * 16 + lrow;
            uint4 pair[2];
            pair[0] = *(const uint4*)&Bs[buf][col][half * 8];        // K 0..7 / 8..15
            pair[1] = *(const uint4*)&Bs[buf][col][16 + half * 8];   // K 16..23 / 24..31
            __builtin_memcpy(&bfrag[nt], pair, 32);
        }
#pragma unroll
        for (int mt = 0; mt < 4; mt++)
#pragma unroll
            for (int nt = 0; nt < 4; nt++)
                acc[mt][nt] = __builtin_amdgcn_wmma_f32_16x16x32_bf16(
                    false, afrag[mt], false, bfrag[nt],
                    (short)0, acc[mt][nt], false, false);

        wait_asynccnt0();       // next-tile async copies complete
        __syncthreads();
        buf ^= 1;
    }

    // store: C/D layout -> VGPR r holds row half*8+r, col = lane%16
#pragma unroll
    for (int mt = 0; mt < 4; mt++) {
#pragma unroll
        for (int nt = 0; nt < 4; nt++) {
            long col = blockN + wn + nt * 16 + lrow;
            float bv = (EPI == 1) ? bias[col] : 0.f;
#pragma unroll
            for (int r = 0; r < 8; r++) {
                long row = blockM + wm + mt * 16 + half * 8 + r;
                float v = acc[mt][nt][r];
                if (EPI == 1) { v += bv; v = v > 0.f ? v : 0.f; }
                C[row * (long)N + col] = v;
            }
        }
    }
}

// ---------------------------------------------------------------------------
// Elementwise kernels (float4-vectorized)
// ---------------------------------------------------------------------------
// h0 = h ; hB = bf16(h)
__global__ void k_copy_h0(const float4* __restrict__ h, float4* __restrict__ h0,
                          uint2* __restrict__ hb, long n4) {
    long i = (long)blockIdx.x * blockDim.x + threadIdx.x;
    long stride = (long)gridDim.x * blockDim.x;
    for (; i < n4; i += stride) { float4 v = h[i]; h0[i] = v; hb[i] = pack4bf(v); }
}

// support = (1-ALPHA)*hi + ALPHA*h0 ; dual-store f32 + bf16
__global__ void k_support(const float4* __restrict__ hi, const float4* __restrict__ h0,
                          float4* __restrict__ sup, uint2* __restrict__ supb, long n4) {
    long i = (long)blockIdx.x * blockDim.x + threadIdx.x;
    long stride = (long)gridDim.x * blockDim.x;
    for (; i < n4; i += stride) {
        float4 a = hi[i], b = h0[i], v;
        v.x = (1.f - ALPHA) * a.x + ALPHA * b.x;
        v.y = (1.f - ALPHA) * a.y + ALPHA * b.y;
        v.z = (1.f - ALPHA) * a.z + ALPHA * b.z;
        v.w = (1.f - ALPHA) * a.w + ALPHA * b.w;
        sup[i] = v; supb[i] = pack4bf(v);
    }
}

__device__ __forceinline__ float actf(float v, int act) {
    if (act == 0) return v > 0.f ? v : 0.f;
    return v > 0.f ? v : (__expf(v) - 1.f);
}

// h = act(theta*y + (1-theta)*sup + h); act 0 = relu, 1 = elu; hB = bf16(h)
__global__ void k_combine(const float4* __restrict__ y, const float4* __restrict__ sup,
                          float4* __restrict__ h, uint2* __restrict__ hb,
                          float theta, int act, long n4) {
    long i = (long)blockIdx.x * blockDim.x + threadIdx.x;
    long stride = (long)gridDim.x * blockDim.x;
    for (; i < n4; i += stride) {
        float4 a = y[i], s = sup[i], c = h[i], v;
        v.x = actf(theta * a.x + (1.f - theta) * s.x + c.x, act);
        v.y = actf(theta * a.y + (1.f - theta) * s.y + c.y, act);
        v.z = actf(theta * a.z + (1.f - theta) * s.z + c.z, act);
        v.w = actf(theta * a.w + (1.f - theta) * s.w + c.w, act);
        h[i] = v; hb[i] = pack4bf(v);
    }
}

// f1[r] = dot(hp[r,:], gat_a[0:1024]) ; f2[r] = dot(hp[r,:], gat_a[1024:2048])
__global__ __launch_bounds__(256)
void k_rowdot2(const float* __restrict__ hp, const float* __restrict__ ga,
               float* __restrict__ f1, float* __restrict__ f2) {
    int wave = threadIdx.x >> 5, lane = threadIdx.x & 31;
    int r = blockIdx.x * 8 + wave;
    const float* row = hp + (long)r * NHID;
    float s1 = 0.f, s2 = 0.f;
    for (int c = lane; c < NHID; c += 32) {
        float v = row[c];
        s1 += v * ga[c];
        s2 += v * ga[NHID + c];
    }
#pragma unroll
    for (int off = 16; off > 0; off >>= 1) {
        s1 += __shfl_xor(s1, off, 32);
        s2 += __shfl_xor(s2, off, 32);
    }
    if (lane == 0) { f1[r] = s1; f2[r] = s2; }
}

// ---------------------------------------------------------------------------
// Fused masked leaky-relu + row softmax -> bf16 attention row.
// Masked-e row (32 KB) staged in LDS: adj and f2 read from HBM exactly once.
// ---------------------------------------------------------------------------
__global__ __launch_bounds__(256)
void k_att_softmax(const float* __restrict__ f1, const float* __restrict__ f2,
                   const int* __restrict__ adj, unsigned short* __restrict__ att) {
    __shared__ float erow[NN];          // 32 KB
    __shared__ float red[256];
    const long i = blockIdx.x;
    const int4*   a4 = (const int4*)(adj + i * (long)NN);
    const float4* v4 = (const float4*)f2;
    const float fi = f1[i];

    float m = -3.0e38f;
    for (int q = threadIdx.x; q < NN / 8; q += 256) {
        int4   m0 = a4[q * 2 + 0], m1 = a4[q * 2 + 1];
        float4 u0 = v4[q * 2 + 0], u1 = v4[q * 2 + 1];
        float e[8];
        e[0] = fi + u0.x; e[1] = fi + u0.y; e[2] = fi + u0.z; e[3] = fi + u0.w;
        e[4] = fi + u1.x; e[5] = fi + u1.y; e[6] = fi + u1.z; e[7] = fi + u1.w;
        int msk[8] = {m0.x, m0.y, m0.z, m0.w, m1.x, m1.y, m1.z, m1.w};
#pragma unroll
        for (int t = 0; t < 8; t++) {
            float v = e[t] > 0.f ? e[t] : 0.2f * e[t];   // leaky_relu(0.2)
            v = msk[t] > 0 ? v : NEGV;
            erow[q * 8 + t] = v;
            m = fmaxf(m, v);
        }
    }
    red[threadIdx.x] = m; __syncthreads();
    for (int s = 128; s > 0; s >>= 1) {
        if (threadIdx.x < s) red[threadIdx.x] = fmaxf(red[threadIdx.x], red[threadIdx.x + s]);
        __syncthreads();
    }
    m = red[0]; __syncthreads();

    float sum = 0.f;
    for (int j = threadIdx.x; j < NN; j += 256) {
        float p = __expf(erow[j] - m);
        erow[j] = p;
        sum += p;
    }
    red[threadIdx.x] = sum; __syncthreads();
    for (int s = 128; s > 0; s >>= 1) {
        if (threadIdx.x < s) red[threadIdx.x] += red[threadIdx.x + s];
        __syncthreads();
    }
    const float inv = 1.f / red[0];

    uint4* orow = (uint4*)(att + i * (long)NN);
    for (int q = threadIdx.x; q < NN / 8; q += 256) {
        float4 p0, p1;
        p0.x = erow[q * 8 + 0] * inv; p0.y = erow[q * 8 + 1] * inv;
        p0.z = erow[q * 8 + 2] * inv; p0.w = erow[q * 8 + 3] * inv;
        p1.x = erow[q * 8 + 4] * inv; p1.y = erow[q * 8 + 5] * inv;
        p1.z = erow[q * 8 + 6] * inv; p1.w = erow[q * 8 + 7] * inv;
        uint2 lo = pack4bf(p0), hi = pack4bf(p1);
        uint4 o; o.x = lo.x; o.y = lo.y; o.z = hi.x; o.w = hi.y;
        orow[q] = o;
    }
}

// out[r] = sigmoid(dot(c1[r,0:64], cls3_W) + cls3_b); one wave per row
__global__ __launch_bounds__(256)
void k_head2(const float* __restrict__ c1, const float* __restrict__ w,
             const float* __restrict__ b, float* __restrict__ out) {
    int wave = threadIdx.x >> 5, lane = threadIdx.x & 31;
    int r = blockIdx.x * 8 + wave;
    const float* row = c1 + (long)r * 128;     // padded width 128, cols 0..63 valid
    int c = lane * 2;
    float s = row[c] * w[c] + row[c + 1] * w[c + 1];
#pragma unroll
    for (int off = 16; off > 0; off >>= 1) s += __shfl_xor(s, off, 32);
    if (lane == 0) out[r] = 1.f / (1.f + __expf(-(s + b[0])));
}

// ---------------------------------------------------------------------------
// Host orchestration
// ---------------------------------------------------------------------------
extern "C" void kernel_launch(void* const* d_in, const int* in_sizes, int n_in,
                              void* d_out, int out_size, void* d_ws, size_t ws_size,
                              hipStream_t stream) {
    const float* x      = (const float*)d_in[0];
    const float* dist   = (const float*)d_in[1];
    const int*   adj    = (const int*)  d_in[2];
    const float* fc_W   = (const float*)d_in[3];
    const float* fc_b   = (const float*)d_in[4];
    const float* conv_W = (const float*)d_in[5];
    const float* gat_W  = (const float*)d_in[6];
    const float* gat_a  = (const float*)d_in[7];
    const float* cls1_W = (const float*)d_in[8];
    const float* cls1_b = (const float*)d_in[9];
    const float* cls3_W = (const float*)d_in[10];
    const float* cls3_b = (const float*)d_in[11];
    float* out = (float*)d_out;

    // ---- workspace carve-out (256B aligned) ----
    size_t off = 0;
    auto alloc = [&](size_t bytes) -> char* {
        char* p = (char*)d_ws + off;
        off += (bytes + 255) & ~(size_t)255;
        return p;
    };
    const long NNL = (long)NN * NN;
    const long NH  = (long)NN * NHID;
    unsigned short* B0     = (unsigned short*)alloc(NNL * 2);   // dist_bf16 / attention_bf16
    float*          hF     = (float*)alloc(NH * 4);
    float*          h0F    = (float*)alloc(NH * 4);
    float*          t1F    = (float*)alloc(NH * 4);             // hi / Y / hp
    float*          supF   = (float*)alloc(NH * 4);
    unsigned short* hB     = (unsigned short*)alloc(NH * 2);    // h, bf16, [NN][NHID]
    unsigned short* hBT    = (unsigned short*)alloc(NH * 2);    // h^T, bf16, [NHID][NN]
    unsigned short* supB   = (unsigned short*)alloc(NH * 2);    // support, bf16 (A operand)
    unsigned short* xpB    = (unsigned short*)alloc((long)NN * 64 * 2);
    unsigned short* fcWT   = (unsigned short*)alloc((long)NHID * 64 * 2);        // [1024][64]
    unsigned short* convWT = (unsigned short*)alloc((long)NLAYERS * NHID * NHID * 2);
    unsigned short* gatWT  = (unsigned short*)alloc((long)NHID * NHID * 2);
    unsigned short* c1WT   = (unsigned short*)alloc(128L * NHID * 2);            // [128][1024]
    float*          biasP  = (float*)alloc(128 * 4);
    float*          c1F    = (float*)alloc((long)NN * 128 * 4);
    float*          f1     = (float*)alloc((long)NN * 4);
    float*          f2     = (float*)alloc((long)NN * 4);
    (void)ws_size; (void)n_in; (void)in_sizes; (void)out_size;

    const dim3 EB(256);
    const dim3 EG(4096);
    const dim3 gemmGrid(NHID / BN, NN / BM);
    const dim3 trGrid(NHID / 32, NN / 32);      // hB [NN][NHID] -> hBT

    // ---- conversions / padding (weights transposed once, to BT layout) ----
    k_f32_to_bf16v<<<EG, EB, 0, stream>>>((const float4*)dist, (uint2*)B0, NNL / 4);
    k_pad_x       <<<EG, EB, 0, stream>>>(x, xpB);
    k_convT       <<<EG, EB, 0, stream>>>(fc_W, fcWT, 64, NHID, NHID, NFEAT, NHID);
    for (int l = 0; l < NLAYERS; l++)
        k_convT   <<<EG, EB, 0, stream>>>(conv_W + (long)l * NHID * NHID,
                                          convWT + (long)l * NHID * NHID,
                                          NHID, NHID, NHID, NHID, NHID);
    k_convT       <<<EG, EB, 0, stream>>>(gat_W, gatWT, NHID, NHID, NHID, NHID, NHID);
    k_convT       <<<EG, EB, 0, stream>>>(cls1_W, c1WT, NHID, 128, 64, NHID, 64);
    k_pad_bias    <<<1, 128, 0, stream>>>(cls1_b, biasP);

    // ---- fc: h = relu(x @ fc_W + b) ----
    k_gemm_bf16<1><<<gemmGrid, 256, 0, stream>>>(xpB, fcWT, hF, NN, NHID, 64, fc_b);
    k_copy_h0<<<EG, EB, 0, stream>>>((const float4*)hF, (float4*)h0F, (uint2*)hB, NH / 4);
    k_transpose_bf16<<<trGrid, 256, 0, stream>>>(hB, hBT, NN, NHID);

    // ---- 7 GCNII layers ----
    for (int l = 0; l < NLAYERS; l++) {
        float theta = fminf(1.f, logf(1.5f / (float)(l + 1) + 1.f));
        // hi = dist @ h        (B = h given transposed as hBT)
        k_gemm_bf16<0><<<gemmGrid, 256, 0, stream>>>(B0, hBT, t1F, NN, NHID, NN, nullptr);
        // support = 0.3*hi + 0.7*h0
        k_support<<<EG, EB, 0, stream>>>((const float4*)t1F, (const float4*)h0F,
                                         (float4*)supF, (uint2*)supB, NH / 4);
        // Y = support @ conv_W[l]
        k_gemm_bf16<0><<<gemmGrid, 256, 0, stream>>>(
            supB, convWT + (long)l * NHID * NHID, t1F, NN, NHID, NHID, nullptr);
        // h = relu(theta*Y + (1-theta)*support + h)
        k_combine<<<EG, EB, 0, stream>>>((const float4*)t1F, (const float4*)supF,
                                         (float4*)hF, (uint2*)hB, theta, 0, NH / 4);
        k_transpose_bf16<<<trGrid, 256, 0, stream>>>(hB, hBT, NN, NHID);
    }

    // ---- GAT attention ----
    // hp = h @ gat_W
    k_gemm_bf16<0><<<gemmGrid, 256, 0, stream>>>(hB, gatWT, t1F, NN, NHID, NHID, nullptr);
    k_rowdot2<<<NN / 8, 256, 0, stream>>>(t1F, gat_a, f1, f2);
    // attention rows (overwrites dist bf16 buffer)
    k_att_softmax<<<NN, 256, 0, stream>>>(f1, f2, adj, B0);
    // hi = attention @ h
    k_gemm_bf16<0><<<gemmGrid, 256, 0, stream>>>(B0, hBT, t1F, NN, NHID, NN, nullptr);
    k_support<<<EG, EB, 0, stream>>>((const float4*)t1F, (const float4*)h0F,
                                     (float4*)supF, (uint2*)supB, NH / 4);
    // Y = support @ gat_W
    k_gemm_bf16<0><<<gemmGrid, 256, 0, stream>>>(supB, gatWT, t1F, NN, NHID, NHID, nullptr);
    {
        float theta9 = fminf(1.f, logf(1.5f / 9.f + 1.f));
        k_combine<<<EG, EB, 0, stream>>>((const float4*)t1F, (const float4*)supF,
                                         (float4*)hF, (uint2*)hB, theta9, 1, NH / 4);  // elu
    }

    // ---- classifier ----
    // c1 = relu(h @ cls1_W + b1)  (N padded to 128)
    k_gemm_bf16<1><<<dim3(128 / BN, NN / BM), 256, 0, stream>>>(
        hB, c1WT, c1F, NN, 128, NHID, biasP);
    k_head2<<<NN / 8, 256, 0, stream>>>(c1F, cls3_W, cls3_b, out);
}
